// MoEClassifier_35957466202332
// MI455X (gfx1250) — matrix operand order repaired and checked by reading, not compiled, not run on hardware
//
#include <hip/hip_runtime.h>

// ---------------------------------------------------------------------------
// MoE classifier for MI455X (gfx1250), wave32 + v_wmma_f32_16x16x32_bf16.
// bf16 N-major weights; 128x128x32 LDS-tiled WMMA GEMM with double-buffered
// GLOBAL_LOAD_ASYNC_TO_LDS_B128 copies (ASYNCcnt); expert loop fused with the
// routing-weighted reduction in the last GEMM epilogue.
// ---------------------------------------------------------------------------

typedef __attribute__((ext_vector_type(16))) __bf16 v16bf;
typedef __attribute__((ext_vector_type(8)))  float  v8f;
typedef __attribute__((ext_vector_type(4)))  int    v4i;

#define B_DIM 8192
#define D_DIM 1024
#define H_DIM 2048
#define C_DIM 1000
#define K_DIM 128
#define E_DIM 8

#define AS1 __attribute__((address_space(1)))
#define AS3 __attribute__((address_space(3)))

#if defined(__has_builtin)
#if __has_builtin(__builtin_amdgcn_global_load_async_to_lds_b128)
#define HAVE_ASYNC_LDS 1
#endif
#endif
#ifndef HAVE_ASYNC_LDS
#define HAVE_ASYNC_LDS 0
#endif

__device__ __forceinline__ void wait_async0() {
#if defined(__has_builtin) && __has_builtin(__builtin_amdgcn_s_wait_asynccnt)
  __builtin_amdgcn_s_wait_asynccnt(0);
#else
  asm volatile("s_wait_asynccnt 0" ::: "memory");
#endif
}

__device__ __forceinline__ void cp16_async(const void* g, void* l) {
#if HAVE_ASYNC_LDS
  __builtin_amdgcn_global_load_async_to_lds_b128(
      (AS1 v4i*)g, (AS3 v4i*)l, 0, 0);
#else
  *(uint4*)l = *(const uint4*)g;
#endif
}

__device__ __forceinline__ unsigned short f2bf(float f) {
  unsigned u = __float_as_uint(f);
  unsigned r = 0x7FFFu + ((u >> 16) & 1u);      // round-to-nearest-even
  return (unsigned short)((u + r) >> 16);
}

// ----------------------------- small utilities -----------------------------

__global__ void moe_cvt_bf16_kernel(const float* __restrict__ in,
                                    unsigned short* __restrict__ out,
                                    long long n) {
  long long i = (long long)blockIdx.x * blockDim.x + threadIdx.x;
  long long stride = (long long)gridDim.x * blockDim.x;
  for (; i < n; i += stride) out[i] = f2bf(in[i]);
}

// in: [batch][R][C] f32 row-major  ->  out: [batch][C][R] bf16 row-major
__global__ void moe_tconv_bf16_kernel(const float* __restrict__ in,
                                      unsigned short* __restrict__ out,
                                      int R, int C, long long total) {
  long long i = (long long)blockIdx.x * blockDim.x + threadIdx.x;
  long long stride = (long long)gridDim.x * blockDim.x;
  long long rc = (long long)R * C;
  for (; i < total; i += stride) {
    long long b = i / rc;
    long long rem = i - b * rc;
    int r = (int)(rem / C);
    int c = (int)(rem - (long long)r * C);
    out[b * rc + (long long)c * R + r] = f2bf(in[i]);
  }
}

__global__ void moe_zero_f32_kernel(float* __restrict__ p, long long n) {
  long long i = (long long)blockIdx.x * blockDim.x + threadIdx.x;
  long long stride = (long long)gridDim.x * blockDim.x;
  for (; i < n; i += stride) p[i] = 0.0f;
}

// ------------------------------ WMMA GEMM ----------------------------------
// C[M,N] = epilogue(A[M,K](bf16) @ Bt[N,K]^T(bf16) + bias[N])
// MODE 0: relu -> bf16 out     MODE 1: linear -> bf16 out
// MODE 2: linear -> f32 out    MODE 3: f32 out += wscale[row,eidx]*(acc+bias)
// GUARD_N: handle N not a multiple of 128 (only the C=1000 GEMM needs it).
template <int MODE, bool GUARD_N>
__global__ __launch_bounds__(256)
void moe_gemm_wmma_kernel(const unsigned short* __restrict__ A,
                          const unsigned short* __restrict__ Bt,
                          const float* __restrict__ bias,
                          void* __restrict__ out,
                          const float* __restrict__ wscale,
                          int M, int N, int K, int eidx) {
  constexpr int BM = 128, BN = 128, BK = 32;
  __shared__ __align__(16) unsigned short sA[2][BM * BK];  // 2 x 8 KB
  __shared__ __align__(16) unsigned short sB[2][BN * BK];  // 2 x 8 KB

  const int tid  = threadIdx.x;
  const int lane = tid & 31;
  const int wid  = tid >> 5;          // 8 waves
  const int wm   = wid >> 1;          // 0..3  -> 32-row strip
  const int wn   = wid & 1;           // 0..1  -> 64-col strip
  const int half = lane >> 4;
  const int lr   = lane & 15;
  const int mBase = blockIdx.y * BM;
  const int nBase = blockIdx.x * BN;

  const int r0 = tid >> 2;            // 0..63 (two passes cover 128 rows)
  const int c0 = (tid & 3) * 8;       // 8 bf16 = 16B per thread

  v8f acc[2][4];
  #pragma unroll
  for (int mi = 0; mi < 2; ++mi)
    #pragma unroll
    for (int ni = 0; ni < 4; ++ni)
      acc[mi][ni] = v8f{0.f, 0.f, 0.f, 0.f, 0.f, 0.f, 0.f, 0.f};

  // tile copy: 16B per thread per row-half; direct global->LDS when async
  // copies are available (no VGPR staging, tracked by ASYNCcnt).
  auto load_tiles = [&](int buf, int kt) {
    cp16_async(&A[(size_t)(mBase + r0) * K + kt + c0], &sA[buf][r0 * BK + c0]);
    cp16_async(&A[(size_t)(mBase + r0 + 64) * K + kt + c0],
               &sA[buf][(r0 + 64) * BK + c0]);
    if (!GUARD_N) {
      cp16_async(&Bt[(size_t)(nBase + r0) * K + kt + c0],
                 &sB[buf][r0 * BK + c0]);
      cp16_async(&Bt[(size_t)(nBase + r0 + 64) * K + kt + c0],
                 &sB[buf][(r0 + 64) * BK + c0]);
    } else {
      uint4 z; z.x = z.y = z.z = z.w = 0u;
      int n1 = nBase + r0;
      if (n1 < N)
        cp16_async(&Bt[(size_t)n1 * K + kt + c0], &sB[buf][r0 * BK + c0]);
      else
        *(uint4*)&sB[buf][r0 * BK + c0] = z;
      int n2 = nBase + r0 + 64;
      if (n2 < N)
        cp16_async(&Bt[(size_t)n2 * K + kt + c0], &sB[buf][(r0 + 64) * BK + c0]);
      else
        *(uint4*)&sB[buf][(r0 + 64) * BK + c0] = z;
    }
  };

  const int nt = K / BK;
  load_tiles(0, 0);
  wait_async0();
  __syncthreads();

  for (int t = 0; t < nt; ++t) {
    const int cur = t & 1;
    if (t + 1 < nt) load_tiles(cur ^ 1, (t + 1) * BK);

    union Frag { unsigned u[8]; v16bf v; };
    Frag a[2], b[4];
    const unsigned* aw = (const unsigned*)sA[cur];   // dword view, 16/row
    const unsigned* bw = (const unsigned*)sB[cur];

    // A operand: lane holds row M=lr; K 0..15 (lane<16)
    //            or 16..31 (lane>=16)
    #pragma unroll
    for (int mi = 0; mi < 2; ++mi) {
      int base = (wm * 32 + mi * 16 + lr) * (BK / 2);
      #pragma unroll
      for (int p = 0; p < 8; ++p)
        a[mi].u[p] = aw[base + half * 8 + p];
    }
    // B operand: lane holds column N=lr; K 0..15 (lane<16) / 16..31 (>=16)
    #pragma unroll
    for (int ni = 0; ni < 4; ++ni) {
      int base = (wn * 64 + ni * 16 + lr) * (BK / 2);
      #pragma unroll
      for (int p = 0; p < 8; ++p)
        b[ni].u[p] = bw[base + half * 8 + p];
    }

    #pragma unroll
    for (int mi = 0; mi < 2; ++mi)
      #pragma unroll
      for (int ni = 0; ni < 4; ++ni)
        acc[mi][ni] = __builtin_amdgcn_wmma_f32_16x16x32_bf16(
            false, a[mi].v, false, b[ni].v, (short)0, acc[mi][ni],
            false, false);

    wait_async0();      // next tile fully in LDS
    __syncthreads();    // all waves done reading current tile
  }

  // Epilogue. C/D layout: VGPR v -> row = v + 8*half, col = lr.
  #pragma unroll
  for (int mi = 0; mi < 2; ++mi) {
    int gr0 = mBase + wm * 32 + mi * 16 + half * 8;
    #pragma unroll
    for (int ni = 0; ni < 4; ++ni) {
      int gc = nBase + wn * 64 + ni * 16 + lr;
      if (GUARD_N && gc >= N) continue;
      float bv = bias[gc];
      #pragma unroll
      for (int v = 0; v < 8; ++v) {
        int grow = gr0 + v;
        float r = acc[mi][ni][v] + bv;
        if (MODE == 0) r = fmaxf(r, 0.0f);
        if (MODE == 0 || MODE == 1) {
          ((unsigned short*)out)[(size_t)grow * N + gc] = f2bf(r);
        } else if (MODE == 2) {
          ((float*)out)[(size_t)grow * N + gc] = r;
        } else {
          float s = wscale[(size_t)grow * E_DIM + eidx];
          float* o = (float*)out + (size_t)grow * N + gc;
          *o += s * r;   // blocks own disjoint elements; sequential experts
        }
      }
    }
  }
}

// ------------------------------- routing -----------------------------------
__global__ void moe_routing_kernel(const float* __restrict__ pk,
                                   const float* __restrict__ keys,
                                   float* __restrict__ weights,
                                   int* __restrict__ topi,
                                   float* __restrict__ sim,
                                   int rows) {
  __shared__ float knorm[E_DIM];
  int t = threadIdx.x;
  if (t < E_DIM) {
    float s = 0.f;
    for (int i = 0; i < K_DIM; ++i) { float v = keys[t * K_DIM + i]; s += v * v; }
    knorm[t] = fmaxf(sqrtf(s), 1e-12f);
  }
  __syncthreads();

  int b = blockIdx.x * blockDim.x + t;
  if (b >= rows) return;
  const float* p = pk + (size_t)b * K_DIM;

  float nrm = 0.f;
  for (int i = 0; i < K_DIM; ++i) nrm += p[i] * p[i];
  nrm = fmaxf(sqrtf(nrm), 1e-12f);

  float s[E_DIM];
  #pragma unroll
  for (int e = 0; e < E_DIM; ++e) {
    const float* kr = keys + e * K_DIM;
    float d = 0.f;
    for (int i = 0; i < K_DIM; ++i) d += p[i] * kr[i];
    s[e] = d / (nrm * knorm[e]);
    sim[(size_t)b * E_DIM + e] = s[e];
  }

  float best = -3.402823466e38f, sec = -3.402823466e38f;
  int bi = 0, si = 0;
  #pragma unroll
  for (int e = 0; e < E_DIM; ++e) {
    if (s[e] > best)     { sec = best; si = bi; best = s[e]; bi = e; }
    else if (s[e] > sec) { sec = s[e]; si = e; }
  }
  float ex = expf(sec - best);
  float w0 = 1.f / (1.f + ex);
  float w1 = ex / (1.f + ex);
  #pragma unroll
  for (int e = 0; e < E_DIM; ++e)
    weights[(size_t)b * E_DIM + e] = (e == bi) ? w0 : ((e == si) ? w1 : 0.f);
  topi[(size_t)b * 2 + 0] = bi;
  topi[(size_t)b * 2 + 1] = si;
}

// ------------------------------- launcher ----------------------------------

extern "C" void kernel_launch(void* const* d_in, const int* in_sizes, int n_in,
                              void* d_out, int out_size, void* d_ws,
                              size_t ws_size, hipStream_t stream) {
  (void)in_sizes; (void)n_in; (void)out_size; (void)ws_size;
  const float* x    = (const float*)d_in[0];
  const float* fw1  = (const float*)d_in[1];
  const float* fb1  = (const float*)d_in[2];
  const float* fw2  = (const float*)d_in[3];
  const float* fb2  = (const float*)d_in[4];
  const float* kw   = (const float*)d_in[5];
  const float* kb   = (const float*)d_in[6];
  const float* keys = (const float*)d_in[7];
  const float* ew1  = (const float*)d_in[8];
  const float* eb1  = (const float*)d_in[9];
  const float* ew2  = (const float*)d_in[10];
  const float* eb2  = (const float*)d_in[11];
  const float* ew3  = (const float*)d_in[12];
  const float* eb3  = (const float*)d_in[13];

  // d_out regions, tuple return order: final, weights, top_i, sim
  float* final_out = (float*)d_out;                               // [B,C]
  float* weights_o = final_out + (size_t)B_DIM * C_DIM;           // [B,E]
  int*   topi_o    = (int*)(weights_o + (size_t)B_DIM * E_DIM);   // [B,2]
  float* sim_o     = (float*)(topi_o + (size_t)B_DIM * 2);        // [B,E]

  // workspace carve (256B aligned)
  char* wp = (char*)d_ws;
  auto carve = [&](size_t bytes) -> void* {
    void* p = (void*)wp;
    wp += (bytes + 255) & ~(size_t)255;
    return p;
  };
  const int Hh = H_DIM / 2;
  unsigned short* xb    = (unsigned short*)carve((size_t)B_DIM * D_DIM * 2);
  unsigned short* fw1t  = (unsigned short*)carve((size_t)H_DIM * D_DIM * 2);
  unsigned short* fw2t  = (unsigned short*)carve((size_t)H_DIM * H_DIM * 2);
  unsigned short* kwt   = (unsigned short*)carve((size_t)K_DIM * H_DIM * 2);
  unsigned short* ew1t  = (unsigned short*)carve((size_t)E_DIM * H_DIM * H_DIM * 2);
  unsigned short* ew2t  = (unsigned short*)carve((size_t)E_DIM * Hh * H_DIM * 2);
  unsigned short* ew3t  = (unsigned short*)carve((size_t)E_DIM * C_DIM * Hh * 2);
  unsigned short* hid   = (unsigned short*)carve((size_t)B_DIM * H_DIM * 2);
  unsigned short* feats = (unsigned short*)carve((size_t)B_DIM * H_DIM * 2);
  unsigned short* h2b   = (unsigned short*)carve((size_t)B_DIM * Hh * 2);
  float*          pk    = (float*)carve((size_t)B_DIM * K_DIM * 4);

  dim3 blk(256);
  auto gs = [](long long n) {
    long long b = (n + 255) / 256;
    if (b > 32768) b = 32768;
    return dim3((unsigned)b);
  };

  // ---- weight / activation conversion to bf16 (weights N-major) ----
  moe_cvt_bf16_kernel<<<gs((long long)B_DIM * D_DIM), blk, 0, stream>>>(
      x, xb, (long long)B_DIM * D_DIM);
  moe_tconv_bf16_kernel<<<gs((long long)D_DIM * H_DIM), blk, 0, stream>>>(
      fw1, fw1t, D_DIM, H_DIM, (long long)D_DIM * H_DIM);
  moe_tconv_bf16_kernel<<<gs((long long)H_DIM * H_DIM), blk, 0, stream>>>(
      fw2, fw2t, H_DIM, H_DIM, (long long)H_DIM * H_DIM);
  moe_tconv_bf16_kernel<<<gs((long long)H_DIM * K_DIM), blk, 0, stream>>>(
      kw, kwt, H_DIM, K_DIM, (long long)H_DIM * K_DIM);
  moe_tconv_bf16_kernel<<<gs((long long)E_DIM * H_DIM * H_DIM), blk, 0, stream>>>(
      ew1, ew1t, H_DIM, H_DIM, (long long)E_DIM * H_DIM * H_DIM);
  moe_tconv_bf16_kernel<<<gs((long long)E_DIM * H_DIM * Hh), blk, 0, stream>>>(
      ew2, ew2t, H_DIM, Hh, (long long)E_DIM * H_DIM * Hh);
  moe_tconv_bf16_kernel<<<gs((long long)E_DIM * Hh * C_DIM), blk, 0, stream>>>(
      ew3, ew3t, Hh, C_DIM, (long long)E_DIM * Hh * C_DIM);

  // ---- feature extractor ----
  {
    dim3 grid(H_DIM / 128, B_DIM / 128);
    moe_gemm_wmma_kernel<0, false><<<grid, blk, 0, stream>>>(
        xb, fw1t, fb1, hid, nullptr, B_DIM, H_DIM, D_DIM, 0);
    moe_gemm_wmma_kernel<1, false><<<grid, blk, 0, stream>>>(
        hid, fw2t, fb2, feats, nullptr, B_DIM, H_DIM, H_DIM, 0);
  }
  // ---- routing projection + routing ----
  {
    dim3 grid(1, B_DIM / 128);
    moe_gemm_wmma_kernel<2, false><<<grid, blk, 0, stream>>>(
        feats, kwt, kb, pk, nullptr, B_DIM, K_DIM, H_DIM, 0);
    moe_routing_kernel<<<B_DIM / 256, blk, 0, stream>>>(
        pk, keys, weights_o, topi_o, sim_o, B_DIM);
  }
  // ---- fused expert loop ----
  moe_zero_f32_kernel<<<gs((long long)B_DIM * C_DIM), blk, 0, stream>>>(
      final_out, (long long)B_DIM * C_DIM);
  for (int e = 0; e < E_DIM; ++e) {
    dim3 g1(H_DIM / 128, B_DIM / 128);
    moe_gemm_wmma_kernel<0, false><<<g1, blk, 0, stream>>>(
        feats, ew1t + (size_t)e * H_DIM * H_DIM, eb1 + (size_t)e * H_DIM,
        hid, nullptr, B_DIM, H_DIM, H_DIM, 0);
    dim3 g2(Hh / 128, B_DIM / 128);
    moe_gemm_wmma_kernel<0, false><<<g2, blk, 0, stream>>>(
        hid, ew2t + (size_t)e * Hh * H_DIM, eb2 + (size_t)e * Hh,
        h2b, nullptr, B_DIM, Hh, H_DIM, 0);
    dim3 g3((C_DIM + 127) / 128, B_DIM / 128);
    moe_gemm_wmma_kernel<3, true><<<g3, blk, 0, stream>>>(
        h2b, ew3t + (size_t)e * C_DIM * Hh, eb3 + (size_t)e * C_DIM,
        final_out, weights_o, B_DIM, C_DIM, Hh, e);
  }
}
